// DistributeSelfAttention_52948356825340
// MI455X (gfx1250) — matrix operand order repaired and checked by reading, not compiled, not run on hardware
//
#include <hip/hip_runtime.h>
#include <hip/hip_bf16.h>

#define NTOK 2048
#define CX   768
#define NH   16
#define DH   48

typedef __attribute__((ext_vector_type(16))) __bf16 v16bf;
typedef __attribute__((ext_vector_type(8)))  float  v8f;
typedef __attribute__((ext_vector_type(4)))  int    v4i;

union Frag { uint4 u[2]; v16bf v; };

#if __has_builtin(__builtin_amdgcn_global_load_async_to_lds_b128)
#define HAS_ASYNC_LDS 1
#else
#define HAS_ASYNC_LDS 0
#endif

__device__ inline void wait_asynccnt0() {
#if __has_builtin(__builtin_amdgcn_s_wait_asynccnt)
  __builtin_amdgcn_s_wait_asynccnt(0);
#else
  asm volatile("s_wait_asynccnt 0x0" ::: "memory");
#endif
}

#if HAS_ASYNC_LDS
__device__ inline void async_b128(const void* g, void* l) {
  __builtin_amdgcn_global_load_async_to_lds_b128(
      (__attribute__((address_space(1))) v4i*)g,
      (__attribute__((address_space(3))) v4i*)l, 0, 0);
}
#endif

__device__ inline unsigned short f2bf(float f) {
  unsigned int u = __float_as_uint(f);
  u += 0x7fffu + ((u >> 16) & 1u);   // round-to-nearest-even
  return (unsigned short)(u >> 16);
}

__device__ inline v8f bwmma(v16bf a, v16bf b, v8f c) {
  return __builtin_amdgcn_wmma_f32_16x16x32_bf16(false, a, false, b, (short)0, c,
                                                 false, false);
}

__device__ inline v8f zero8() {
  v8f z = {0.f, 0.f, 0.f, 0.f, 0.f, 0.f, 0.f, 0.f};
  return z;
}

// ---------------------------------------------------------------- LayerNorm
__global__ __launch_bounds__(256)
void ln_kernel(const float* __restrict__ x, const float* __restrict__ g,
               const float* __restrict__ b, unsigned short* __restrict__ xn) {
  const int row = blockIdx.x;
  const int t = threadIdx.x;
  const float v0 = x[(size_t)row * CX + t];
  const float v1 = x[(size_t)row * CX + 256 + t];
  const float v2 = x[(size_t)row * CX + 512 + t];
  __shared__ float r1[256], r2[256];
  r1[t] = v0 + v1 + v2;
  r2[t] = v0 * v0 + v1 * v1 + v2 * v2;
  __syncthreads();
  for (int off = 128; off > 0; off >>= 1) {
    if (t < off) { r1[t] += r1[t + off]; r2[t] += r2[t + off]; }
    __syncthreads();
  }
  const float mu  = r1[0] * (1.0f / CX);
  const float var = r2[0] * (1.0f / CX) - mu * mu;
  const float rs  = rsqrtf(var + 1e-5f);
  xn[(size_t)row * CX + t]       = f2bf((v0 - mu) * rs * g[t]       + b[t]);
  xn[(size_t)row * CX + 256 + t] = f2bf((v1 - mu) * rs * g[256 + t] + b[256 + t]);
  xn[(size_t)row * CX + 512 + t] = f2bf((v2 - mu) * rs * g[512 + t] + b[512 + t]);
}

// ---------------------------------------------------------------- weights -> bf16
__global__ __launch_bounds__(256)
void convw_kernel(const float* __restrict__ w0, const float* __restrict__ w1,
                  const float* __restrict__ w2, const float* __restrict__ w3,
                  const float* __restrict__ w4, unsigned short* __restrict__ dst) {
  const int i = blockIdx.x * 256 + threadIdx.x;
  const float* src = (blockIdx.y == 0) ? w0 : (blockIdx.y == 1) ? w1 :
                     (blockIdx.y == 2) ? w2 : (blockIdx.y == 3) ? w3 : w4;
  dst[(size_t)blockIdx.y * (CX * CX) + i] = f2bf(src[i]);
}

// ---------------------------------------------------------------- GEMM + epilogue
// C[2048x768] = A_bf16 @ W_bf16 ; block tile 64x64, 4 waves x (32x32), K-step 32.
#define GS_A 40   // padded LDS row stride (halfwords); 80B = 5*16B (b128-aligned)
#define GS_B 40

__global__ __launch_bounds__(128)
void gemm_ep_kernel(const unsigned short* __restrict__ A,
                    const unsigned short* __restrict__ Wb,
                    const float* __restrict__ bq,
                    unsigned short* __restrict__ Qb,
                    unsigned short* __restrict__ Kb,
                    unsigned short* __restrict__ Vb,
                    float* __restrict__ gate,
                    float* __restrict__ outF,
                    int modeSel) {
  __shared__ __align__(16) unsigned short Alds[64 * GS_A];
  __shared__ __align__(16) unsigned short Blds[64 * GS_B];

  const int tid  = threadIdx.x;
  const int wave = tid >> 5;
  const int lane = tid & 31;
  const int half = lane >> 4;
  const int l16  = lane & 15;

  const int n0 = blockIdx.x * 64;
  const int m0 = blockIdx.y * 64;
  const int mode = (modeSel < 0) ? (int)blockIdx.z : modeSel;
  const unsigned short* W =
      (modeSel < 0) ? (Wb + (size_t)blockIdx.z * (CX * CX)) : Wb;

  const int wm = (wave >> 1) * 32;
  const int wn = (wave & 1) * 32;

  v8f acc[2][2];
  acc[0][0] = zero8(); acc[0][1] = zero8();
  acc[1][0] = zero8(); acc[1][1] = zero8();

  for (int k0 = 0; k0 < CX; k0 += 32) {
    __syncthreads();
#if HAS_ASYNC_LDS
    // async stage A tile: 64 rows x 4 b128 segs (16B) per row
#pragma unroll
    for (int j = 0; j < 2; ++j) {
      int idx = tid + j * 128;
      int r = idx >> 2, seg = idx & 3;
      const char* src = (const char*)A + ((size_t)(m0 + r) * CX + k0) * 2 + seg * 16;
      char*       dst = (char*)Alds + r * (GS_A * 2) + seg * 16;
      async_b128(src, dst);
    }
#else
    const unsigned int* Au = (const unsigned int*)A;
#pragma unroll
    for (int j = 0; j < 8; ++j) {
      int w = tid + j * 128;
      int r = w >> 4, cw = w & 15;
      unsigned int val = Au[((size_t)(m0 + r) * CX + k0) / 2 + cw];
      *(unsigned int*)&Alds[r * GS_A + cw * 2] = val;
    }
#endif
    // stage B tile transposed: Blds[n][k] = W[k0+k][n0+n]
#pragma unroll
    for (int j = 0; j < 16; ++j) {
      int w = tid + j * 128;
      int kk = w >> 6, nn = w & 63;
      Blds[nn * GS_B + kk] = W[(size_t)(k0 + kk) * CX + n0 + nn];
    }
#if HAS_ASYNC_LDS
    wait_asynccnt0();
#endif
    __syncthreads();

    Frag a[2], b[2];
#pragma unroll
    for (int mt = 0; mt < 2; ++mt) {
      int row = wm + mt * 16 + l16;
      a[mt].u[0] = *(const uint4*)&Alds[row * GS_A + half * 16];
      a[mt].u[1] = *(const uint4*)&Alds[row * GS_A + half * 16 + 8];
    }
#pragma unroll
    for (int nt = 0; nt < 2; ++nt) {
      int col = wn + nt * 16 + l16;
      b[nt].u[0] = *(const uint4*)&Blds[col * GS_B + half * 16];
      b[nt].u[1] = *(const uint4*)&Blds[col * GS_B + half * 16 + 8];
    }
#pragma unroll
    for (int mt = 0; mt < 2; ++mt)
#pragma unroll
      for (int nt = 0; nt < 2; ++nt)
        acc[mt][nt] = bwmma(a[mt].v, b[nt].v, acc[mt][nt]);
  }

  const float qscale = 0.14433756729740643f;  // 1/sqrt(48)
#pragma unroll
  for (int mt = 0; mt < 2; ++mt) {
#pragma unroll
    for (int nt = 0; nt < 2; ++nt) {
      const int col = n0 + wn + nt * 16 + l16;
#pragma unroll
      for (int r = 0; r < 8; ++r) {
        const int row = m0 + wm + mt * 16 + r + 8 * half;
        float v = acc[mt][nt][r];
        if (mode == 0) {            // Q: +bias, fold softmax scale
          v = (v + bq[col]) * qscale;
          int hh = col / DH, dd = col % DH;
          Qb[((size_t)hh * NTOK + row) * DH + dd] = f2bf(v);
        } else if (mode == 1) {     // K
          int hh = col / DH, dd = col % DH;
          Kb[((size_t)hh * NTOK + row) * DH + dd] = f2bf(v);
        } else if (mode == 2) {     // V
          int hh = col / DH, dd = col % DH;
          Vb[((size_t)hh * NTOK + row) * DH + dd] = f2bf(v);
        } else if (mode == 3) {     // gate = sigmoid (fast rcp)
          gate[(size_t)row * CX + col] =
              __builtin_amdgcn_rcpf(1.0f + __expf(-v));
        } else {                    // final output
          outF[(size_t)row * CX + col] = v;
        }
      }
    }
  }
}

// ---------------------------------------------------------------- flash attention
#define KS_S 72   // 144B row stride = 9*16B (b128-aligned)
#define QS_S 72
#define VS_S 40
#define PS_S 40

__global__ __launch_bounds__(128)
void attn_kernel(const unsigned short* __restrict__ Qb,
                 const unsigned short* __restrict__ Kb,
                 const unsigned short* __restrict__ Vb,
                 const float* __restrict__ pl,
                 const unsigned char* __restrict__ mask,
                 const float* __restrict__ gate,
                 float* __restrict__ waOut,
                 unsigned short* __restrict__ waBf) {
  __shared__ __align__(16) unsigned short Ks[32 * KS_S];
  __shared__ __align__(16) unsigned short Vst[DH * VS_S];
  __shared__ __align__(16) unsigned short QsAll[4 * 16 * QS_S];
  __shared__ __align__(16) unsigned short PsAll[4 * 16 * PS_S];

  const int tid  = threadIdx.x;
  const int wave = tid >> 5;
  const int lane = tid & 31;
  const int half = lane >> 4;
  const int l16  = lane & 15;
  const int h    = blockIdx.y;
  const int q0   = blockIdx.x * 64;
  const int wq0  = q0 + wave * 16;

  unsigned short* Qs = QsAll + wave * 16 * QS_S;
  unsigned short* Ps = PsAll + wave * 16 * PS_S;

  // stage this wave's 16 query rows, d padded 48 -> 64 with zeros
  for (int idx = lane; idx < 16 * 64; idx += 32) {
    int m = idx >> 6, d = idx & 63;
    unsigned short v = 0;
    if (d < DH) v = Qb[((size_t)h * NTOK + wq0 + m) * DH + d];
    Qs[m * QS_S + d] = v;
  }
  asm volatile("s_wait_dscnt 0" ::: "memory");

  Frag aq[2];
#pragma unroll
  for (int kk = 0; kk < 2; ++kk) {
    aq[kk].u[0] = *(const uint4*)&Qs[l16 * QS_S + kk * 32 + half * 16];
    aq[kk].u[1] = *(const uint4*)&Qs[l16 * QS_S + kk * 32 + half * 16 + 8];
  }

  float m_run[8], l_run[8];
#pragma unroll
  for (int r = 0; r < 8; ++r) { m_run[r] = -3.0e38f; l_run[r] = 0.0f; }
  v8f o0 = zero8(), o1 = zero8(), o2 = zero8();

  for (int kb = 0; kb < NTOK / 32; ++kb) {
    const int key0 = kb * 32;
    __syncthreads();
#if HAS_ASYNC_LDS
    // async stage K tile: 32 rows x 6 b128 segs of real data (96B = DH bf16)
    for (int idx = tid; idx < 32 * 6; idx += 128) {
      int ky = idx / 6, seg = idx % 6;
      const char* src =
          (const char*)Kb + ((size_t)h * NTOK + key0 + ky) * (DH * 2) + seg * 16;
      char* dst = (char*)Ks + ky * (KS_S * 2) + seg * 16;
      async_b128(src, dst);
    }
    // zero-fill pad segs 6..7 (d = 48..63)
    {
      uint4 z = {0u, 0u, 0u, 0u};
      for (int idx = tid; idx < 32 * 2; idx += 128) {
        int ky = idx >> 1, seg = 6 + (idx & 1);
        *(uint4*)&Ks[ky * KS_S + seg * 8] = z;
      }
    }
#else
    for (int idx = tid; idx < 32 * 64; idx += 128) {
      int ky = idx >> 6, d = idx & 63;
      unsigned short v = 0;
      if (d < DH) v = Kb[((size_t)h * NTOK + key0 + ky) * DH + d];
      Ks[ky * KS_S + d] = v;
    }
#endif
    // V tile transposed: Vst[d][key]
    for (int idx = tid; idx < 32 * DH; idx += 128) {
      int ky = idx / DH, d = idx % DH;
      Vst[d * VS_S + ky] = Vb[((size_t)h * NTOK + key0 + ky) * DH + d];
    }
#if HAS_ASYNC_LDS
    wait_asynccnt0();
#endif
    __syncthreads();

    // S = Q K^T  (16 q-rows x 32 keys), K-dim 64 (padded)
    v8f s0 = zero8(), s1 = zero8();
#pragma unroll
    for (int kk = 0; kk < 2; ++kk) {
      const int koff = kk * 32 + half * 16;
      Frag b0, b1;
      b0.u[0] = *(const uint4*)&Ks[l16 * KS_S + koff];
      b0.u[1] = *(const uint4*)&Ks[l16 * KS_S + koff + 8];
      b1.u[0] = *(const uint4*)&Ks[(16 + l16) * KS_S + koff];
      b1.u[1] = *(const uint4*)&Ks[(16 + l16) * KS_S + koff + 8];
      s0 = bwmma(aq[kk].v, b0.v, s0);
      s1 = bwmma(aq[kk].v, b1.v, s1);
    }

    // bias + online softmax (rows live per (r, lane-half); reduce within half)
#pragma unroll
    for (int r = 0; r < 8; ++r) {
      const int qrow = wq0 + r + 8 * half;
      const size_t pbase = ((size_t)h * NTOK + qrow) * NTOK + key0;
      const size_t mbase = (size_t)qrow * NTOK + key0;
      if (kb + 2 < NTOK / 32)
        __builtin_prefetch((const void*)&pl[pbase + 64 + l16], 0, 0);
      float s0v = s0[r] + pl[pbase + l16]      + (mask[mbase + l16]      ? 0.f : -1e9f);
      float s1v = s1[r] + pl[pbase + 16 + l16] + (mask[mbase + 16 + l16] ? 0.f : -1e9f);
      float mx = fmaxf(s0v, s1v);
#pragma unroll
      for (int off = 8; off >= 1; off >>= 1) mx = fmaxf(mx, __shfl_xor(mx, off, 32));
      const float nm = fmaxf(m_run[r], mx);
      const float alpha = __expf(m_run[r] - nm);
      const float p0 = __expf(s0v - nm);
      const float p1 = __expf(s1v - nm);
      float rs = p0 + p1;
#pragma unroll
      for (int off = 8; off >= 1; off >>= 1) rs += __shfl_xor(rs, off, 32);
      l_run[r] = l_run[r] * alpha + rs;
      m_run[r] = nm;
      o0[r] *= alpha; o1[r] *= alpha; o2[r] *= alpha;
      Ps[(r + 8 * half) * PS_S + l16]      = f2bf(p0);
      Ps[(r + 8 * half) * PS_S + 16 + l16] = f2bf(p1);
    }
    asm volatile("s_wait_dscnt 0" ::: "memory");

    // O += P V  (A = P 16x32 via LDS reshape, B = V^T tiles)
    Frag ap;
    ap.u[0] = *(const uint4*)&Ps[l16 * PS_S + half * 16];
    ap.u[1] = *(const uint4*)&Ps[l16 * PS_S + half * 16 + 8];
    {
      Frag bv;
      bv.u[0] = *(const uint4*)&Vst[(0 + l16) * VS_S + half * 16];
      bv.u[1] = *(const uint4*)&Vst[(0 + l16) * VS_S + half * 16 + 8];
      o0 = bwmma(ap.v, bv.v, o0);
      bv.u[0] = *(const uint4*)&Vst[(16 + l16) * VS_S + half * 16];
      bv.u[1] = *(const uint4*)&Vst[(16 + l16) * VS_S + half * 16 + 8];
      o1 = bwmma(ap.v, bv.v, o1);
      bv.u[0] = *(const uint4*)&Vst[(32 + l16) * VS_S + half * 16];
      bv.u[1] = *(const uint4*)&Vst[(32 + l16) * VS_S + half * 16 + 8];
      o2 = bwmma(ap.v, bv.v, o2);
    }
  }

  // normalize, gate, store wa (fp32 to d_out, bf16 copy for final GEMM)
#pragma unroll
  for (int r = 0; r < 8; ++r) {
    const int row = wq0 + r + 8 * half;
    const float inv = __builtin_amdgcn_rcpf(l_run[r]);
#pragma unroll
    for (int dt = 0; dt < 3; ++dt) {
      const int col = h * DH + dt * 16 + l16;
      float v = (dt == 0 ? o0[r] : dt == 1 ? o1[r] : o2[r]) * inv;
      v *= gate[(size_t)row * CX + col];
      waOut[(size_t)row * CX + col] = v;
      waBf[(size_t)row * CX + col]  = f2bf(v);
    }
  }
}

// ---------------------------------------------------------------- launcher
extern "C" void kernel_launch(void* const* d_in, const int* in_sizes, int n_in,
                              void* d_out, int out_size, void* d_ws, size_t ws_size,
                              hipStream_t stream) {
  (void)in_sizes; (void)n_in; (void)out_size; (void)ws_size;

  const float*         x    = (const float*)d_in[0];
  const unsigned char* mask = (const unsigned char*)d_in[1];
  const float*         pl   = (const float*)d_in[2];
  const float*         ln_g = (const float*)d_in[3];
  const float*         ln_b = (const float*)d_in[4];
  const float*         Wq   = (const float*)d_in[5];
  const float*         bq   = (const float*)d_in[6];
  const float*         Wk   = (const float*)d_in[7];
  const float*         Wv   = (const float*)d_in[8];
  const float*         Wg   = (const float*)d_in[9];
  const float*         Wo   = (const float*)d_in[10];

  char* ws = (char*)d_ws;
  size_t off = 0;
  auto alloc = [&](size_t bytes) -> char* {
    char* p = ws + off;
    off = (off + bytes + 255) & ~(size_t)255;
    return p;
  };
  unsigned short* xn_bf = (unsigned short*)alloc((size_t)NTOK * CX * 2);
  unsigned short* W_bf  = (unsigned short*)alloc((size_t)5 * CX * CX * 2);
  unsigned short* Qbuf  = (unsigned short*)alloc((size_t)NH * NTOK * DH * 2);
  unsigned short* Kbuf  = (unsigned short*)alloc((size_t)NH * NTOK * DH * 2);
  unsigned short* Vbuf  = (unsigned short*)alloc((size_t)NH * NTOK * DH * 2);
  float*          gate  = (float*)alloc((size_t)NTOK * CX * 4);
  unsigned short* wa_bf = (unsigned short*)alloc((size_t)NTOK * CX * 2);

  float* outWa = (float*)d_out;                       // wa  (2048 x 768)
  float* outO  = outWa + (size_t)NTOK * CX;           // out (2048 x 768)

  ln_kernel<<<NTOK, 256, 0, stream>>>(x, ln_g, ln_b, xn_bf);
  convw_kernel<<<dim3(CX * CX / 256, 5), 256, 0, stream>>>(Wq, Wk, Wv, Wg, Wo, W_bf);

  // Q, K, V, gate projections (blockIdx.z selects matrix + epilogue)
  gemm_ep_kernel<<<dim3(CX / 64, NTOK / 64, 4), 128, 0, stream>>>(
      xn_bf, W_bf, bq, Qbuf, Kbuf, Vbuf, gate, nullptr, -1);

  attn_kernel<<<dim3(NTOK / 64, NH), 128, 0, stream>>>(
      Qbuf, Kbuf, Vbuf, pl, mask, gate, outWa, wa_bf);

  // out = wa_gated @ Wo
  gemm_ep_kernel<<<dim3(CX / 64, NTOK / 64, 1), 128, 0, stream>>>(
      wa_bf, W_bf + (size_t)4 * CX * CX, nullptr, nullptr, nullptr, nullptr,
      nullptr, outO, 4);
}
